// GNNEncoder_43001212567825
// MI455X (gfx1250) — compile-verified, matrix operand order checked by reading, not run on hardware
//
#include <hip/hip_runtime.h>
#include <hip/hip_bf16.h>
#include <math.h>

#define BB   4
#define VV   200
#define HH   128
#define LL   3
#define EPSC 1e-5f
#define NROWS_EDGE (BB*VV*VV)          // 160000
#define NROWS_NODE (BB*VV)             // 800

typedef __bf16 bf16_t;
typedef bf16_t v16bf __attribute__((ext_vector_type(16)));
typedef bf16_t v8bf  __attribute__((ext_vector_type(8)));
typedef float  v8f   __attribute__((ext_vector_type(8)));

union BFrag { v16bf v; v8bf h[2]; };
union F8    { v8f   v; float  f[8]; };

union WEU {                 // weights (pre-transposed) / f32 edge tile share LDS
  __hip_bfloat16 w[128][136];
  float          e[64][132];
};

__device__ __forceinline__ float sigmoidf_(float x) { return 1.f / (1.f + __expf(-x)); }

__device__ __forceinline__ v8f zero8() {
  F8 z;
#pragma unroll
  for (int i = 0; i < 8; ++i) z.f[i] = 0.f;
  return z.v;
}

// ---------------------------------------------------------------------------
// CDNA5 async global->LDS copy path (GLOBAL_LOAD_ASYNC_TO_LDS_B128, ASYNCcnt).
// LDS byte address = low 32 bits of the flat shared pointer (LDS aperture).
// ---------------------------------------------------------------------------
__device__ __forceinline__ void async_b128(void* lds, const void* g) {
  asm volatile("global_load_async_to_lds_b128 %0, %1, off"
               :: "v"((uint32_t)(uintptr_t)lds), "v"(g)
               : "memory");
}
__device__ __forceinline__ void async_wait0() {
  asm volatile("s_wait_asynccnt 0" ::: "memory");
}
__device__ __forceinline__ void async_wait4() {       // allow 4 newest in flight
  asm volatile("s_wait_asynccnt 4" ::: "memory");
}

// weight copy: global wbf is PRE-TRANSPOSED [n][k]; straight 16B-chunk copy.
// 2048 chunks -> 8 async instructions per thread.
__device__ __forceinline__ void async_copy_w(const __hip_bfloat16* __restrict__ wg,
                                             __hip_bfloat16 (*Ws)[136], int tid)
{
#pragma unroll
  for (int it = 0; it < 8; ++it) {
    const int c = tid + it * 256;              // 0..2047
    const int n = c >> 4, k8 = (c & 15) << 3;
    async_b128(&Ws[n][k8], wg + n * 128 + k8);
  }
}

// 64x128 bf16 activation tile copy: 1024 chunks -> 4 async instr per thread.
__device__ __forceinline__ void async_copy_tile(const __hip_bfloat16* __restrict__ g,
                                                __hip_bfloat16 (*T)[136], int tid)
{
#pragma unroll
  for (int it = 0; it < 4; ++it) {
    const int c = tid + it * 256;              // 0..1023
    const int m = c >> 4, n8 = (c & 15) << 3;
    async_b128(&T[m][n8], g + m * 128 + n8);
  }
}

// ---------------------------------------------------------------------------
// C(64x128,f32) += A(64x128 bf16, LDS As) * W(128x128 bf16, LDS Ws[n][k]).
// Wave `ns/16` owns N slab [ns,ns+16); acc[mb] = rows [16mb,16mb+16).
// 16-bit A layout: lane&15 = row, lane>>4 selects K sub-blocks.
// ---------------------------------------------------------------------------
__device__ __forceinline__ void tile_gemm(const __hip_bfloat16 (*As)[136],
                                          const __hip_bfloat16 (*Ws)[136],
                                          int lane, int ns, v8f* acc)
{
  const int lm  = lane & 15;
  const int kb8 = (lane >> 4) << 3;
  const bf16_t* wr = (const bf16_t*)&Ws[ns + lm][0];
#pragma unroll
  for (int kk = 0; kk < 128; kk += 32) {
    BFrag bfr;
    bfr.h[0] = *(const v8bf*)(wr + kk + kb8);
    bfr.h[1] = *(const v8bf*)(wr + kk + kb8 + 16);
#pragma unroll
    for (int mb = 0; mb < 4; ++mb) {
      const bf16_t* ar = (const bf16_t*)&As[mb * 16 + lm][0];
      BFrag afr;
      afr.h[0] = *(const v8bf*)(ar + kk + kb8);
      afr.h[1] = *(const v8bf*)(ar + kk + kb8 + 16);
      acc[mb] = __builtin_amdgcn_wmma_f32_16x16x32_bf16(
          false, afr.v, false, bfr.v, (short)0, acc[mb], false, false);
    }
  }
}

// ---------------------------------------------------------------------------
// Sine-embed a 64-row tile of the graph and multiply by edge_w -> bf16 e tensor
// (async weight fill overlaps the sin/cos A-tile construction)
// ---------------------------------------------------------------------------
__global__ __launch_bounds__(256)
void k_edge_embed(const float* __restrict__ graph,
                  const __hip_bfloat16* __restrict__ wedge,   // transposed [n][k]
                  const float* __restrict__ edge_b,
                  __hip_bfloat16* __restrict__ eout)
{
  __shared__ __align__(16) __hip_bfloat16 As[64][136];
  __shared__ __align__(16) __hip_bfloat16 Ws[128][136];
  const int tid = threadIdx.x, lane = tid & 31, wv = tid >> 5;
  const int r0  = blockIdx.x * 64;
  const float LOG1E4 = 9.2103403719761836f;   // ln(10000)

  async_copy_w(wedge, Ws, tid);

  for (int idx = tid; idx < 64 * 128; idx += 256) {
    const int m = idx >> 7, n = idx & 127;
    const float val = graph[r0 + m];
    const float p   = val * __expf(-LOG1E4 * (float)(2 * (n >> 1)) * (1.f / 128.f));
    As[m][n] = __float2bfloat16((n & 1) ? __cosf(p) : __sinf(p));
  }
  async_wait0();
  __syncthreads();

  v8f acc[4];
#pragma unroll
  for (int mb = 0; mb < 4; ++mb) acc[mb] = zero8();
  tile_gemm(As, Ws, lane, wv * 16, acc);

  const int n = wv * 16 + (lane & 15);
  const float bias = edge_b[n];
#pragma unroll
  for (int mb = 0; mb < 4; ++mb) {
    F8 u; u.v = acc[mb];
#pragma unroll
    for (int r = 0; r < 8; ++r) {
      const int m = mb * 16 + ((lane >> 4) << 3) + r;
      eout[(size_t)(r0 + m) * HH + n] = __float2bfloat16(u.f[r] + bias);
    }
  }
}

// ---------------------------------------------------------------------------
// Node-side GEMMs: Uh/Vh/Ah/Bh = h @ W + b   (blockIdx.y selects matrix)
// ---------------------------------------------------------------------------
__global__ __launch_bounds__(256)
void k_node_gemm(const float* __restrict__ hbuf,
                 const __hip_bfloat16* __restrict__ wlayer,   // 7 slots, use 0..3
                 const float* __restrict__ ub, const float* __restrict__ vb,
                 const float* __restrict__ ab, const float* __restrict__ bb,
                 float* __restrict__ uvab)
{
  __shared__ __align__(16) __hip_bfloat16 As[64][136];
  __shared__ __align__(16) __hip_bfloat16 Ws[128][136];
  const int tid = threadIdx.x, lane = tid & 31, wv = tid >> 5;
  const int mat = blockIdx.y;
  const int r0  = blockIdx.x * 64;

  async_copy_w(wlayer + (size_t)mat * HH * HH, Ws, tid);

  for (int idx = tid; idx < 64 * 128; idx += 256) {
    const int m = idx >> 7, n = idx & 127;
    const int r = r0 + m;
    As[m][n] = __float2bfloat16(r < NROWS_NODE ? hbuf[(size_t)r * HH + n] : 0.f);
  }
  async_wait0();
  __syncthreads();

  v8f acc[4];
#pragma unroll
  for (int mb = 0; mb < 4; ++mb) acc[mb] = zero8();
  tile_gemm(As, Ws, lane, wv * 16, acc);

  const float* bias = (mat == 0) ? ub : (mat == 1) ? vb : (mat == 2) ? ab : bb;
  const int n = wv * 16 + (lane & 15);
  const float bv = bias[n];
  float* out = uvab + (size_t)mat * NROWS_NODE * HH;
#pragma unroll
  for (int mb = 0; mb < 4; ++mb) {
    F8 u; u.v = acc[mb];
#pragma unroll
    for (int r = 0; r < 8; ++r) {
      const int m = mb * 16 + ((lane >> 4) << 3) + r;
      const int rg = r0 + m;
      if (rg < NROWS_NODE) out[(size_t)rg * HH + n] = u.f[r] + bv;
    }
  }
}

// ---------------------------------------------------------------------------
// Fused per-layer edge kernel.  One block = fixed (b,i), 64 j-rows, all 128 ch.
// Async double-buffered pipeline:
//   issue As0<-e_noisy, Ws<-C_w, As1<-e_cond  (async to LDS)
//   wait(first 12) -> GEMM1 while e_cond tile lands
//   Ws<-D_w async -> GEMM2
//   e = acc + C_b + D_b + Ah[j] + Bh[i]  -> LDS f32 tile (reuses Ws space)
//   agg atomics, LN1+relu+temb, LN2+SiLU -> bf16 act tile (As0)
//   Ws<-po_w async -> GEMM3 -> residual store
// ---------------------------------------------------------------------------
__global__ __launch_bounds__(256)
void k_edge_layer(__hip_bfloat16* __restrict__ eN,
                  const __hip_bfloat16* __restrict__ eC,
                  const __hip_bfloat16* __restrict__ wC,
                  const __hip_bfloat16* __restrict__ wD,
                  const __hip_bfloat16* __restrict__ wpo,
                  const float* __restrict__ Cb, const float* __restrict__ Db,
                  const float* __restrict__ pob,
                  const float* __restrict__ lnEg, const float* __restrict__ lnEb,
                  const float* __restrict__ poLg, const float* __restrict__ poLb,
                  const float* __restrict__ uvab, float* __restrict__ agg,
                  const float* __restrict__ tlqv)
{
  __shared__ __align__(16) __hip_bfloat16 As0[64][136];
  __shared__ __align__(16) __hip_bfloat16 As1[64][136];
  __shared__ __align__(16) WEU WE;

  const int tid = threadIdx.x, lane = tid & 31, wv = tid >> 5;
  const int jt = blockIdx.x, i = blockIdx.y, b = blockIdx.z;
  const int j0 = jt * 64;
  const int nj = min(64, VV - j0);
  const size_t rowbase = ((size_t)(b * VV + i)) * VV + j0;
  const float* Vh = uvab + (size_t)1 * NROWS_NODE * HH;
  const float* Ah = uvab + (size_t)2 * NROWS_NODE * HH;
  const float* Bh = uvab + (size_t)3 * NROWS_NODE * HH;

  // ---- async issue: As0 (4/thr), Ws=C_w (8/thr), As1 (4/thr)
  async_copy_tile(eN + rowbase * HH, As0, tid);
  async_copy_w(wC, WE.w, tid);
  async_copy_tile(eC + rowbase * HH, As1, tid);
  async_wait4();            // in-order retire: As0 + C_w complete, As1 may fly
  __syncthreads();

  v8f acc[4];
#pragma unroll
  for (int mb = 0; mb < 4; ++mb) acc[mb] = zero8();
  tile_gemm(As0, WE.w, lane, wv * 16, acc);       // Ce, overlapped with As1 load
  __syncthreads();

  async_copy_w(wD, WE.w, tid);
  async_wait0();
  __syncthreads();
  tile_gemm(As1, WE.w, lane, wv * 16, acc);       // + De
  __syncthreads();

  // ---- e tile -> LDS f32 (reuse weight LDS)
  {
    const int n = wv * 16 + (lane & 15);
    const float cst = Cb[n] + Db[n] + Bh[(size_t)(b * VV + i) * HH + n];
#pragma unroll
    for (int mb = 0; mb < 4; ++mb) {
      F8 u; u.v = acc[mb];
#pragma unroll
      for (int r = 0; r < 8; ++r) {
        const int m = mb * 16 + ((lane >> 4) << 3) + r;
        WE.e[m][n] = (m < nj)
            ? u.f[r] + cst + Ah[(size_t)(b * VV + j0 + m) * HH + n]
            : 0.f;
      }
    }
  }
  __syncthreads();

  // ---- agg pass: sum_j sigmoid(e[j,n]) * Vh[b,j,n]   (column-parallel)
  {
    const int n = tid & 127, half = tid >> 7;
    float s = 0.f;
    const int mhi = min(half * 32 + 32, nj);
    for (int m = half * 32; m < mhi; ++m)
      s += sigmoidf_(WE.e[m][n]) * Vh[(size_t)(b * VV + j0 + m) * HH + n];
    unsafeAtomicAdd(&agg[(size_t)(b * VV + i) * HH + n], s);
  }
  __syncthreads();

  // ---- LN1 + relu + time-embed injection (row-parallel: 4 lanes / row)
  {
    const int m = tid >> 2;
    const int n0 = (tid & 3) * 32;
    float s = 0.f, ss = 0.f;
    for (int k = 0; k < 32; ++k) { const float v = WE.e[m][n0 + k]; s += v; ss += v * v; }
    s  += __shfl_xor(s, 1);  s  += __shfl_xor(s, 2);
    ss += __shfl_xor(ss, 1); ss += __shfl_xor(ss, 2);
    const float mean = s * (1.f / 128.f);
    const float rs = rsqrtf(ss * (1.f / 128.f) - mean * mean + EPSC);
    for (int k = 0; k < 32; ++k) {
      const int n = n0 + k;
      const float v = (WE.e[m][n] - mean) * rs * lnEg[n] + lnEb[n];
      WE.e[m][n] = fmaxf(v, 0.f) + tlqv[b * HH + n];
    }
  }
  __syncthreads();

  // ---- LN2 + SiLU -> bf16 act tile (reuse As0)
  {
    const int m = tid >> 2;
    const int n0 = (tid & 3) * 32;
    float s = 0.f, ss = 0.f;
    for (int k = 0; k < 32; ++k) { const float v = WE.e[m][n0 + k]; s += v; ss += v * v; }
    s  += __shfl_xor(s, 1);  s  += __shfl_xor(s, 2);
    ss += __shfl_xor(ss, 1); ss += __shfl_xor(ss, 2);
    const float mean = s * (1.f / 128.f);
    const float rs = rsqrtf(ss * (1.f / 128.f) - mean * mean + EPSC);
    for (int k = 0; k < 32; ++k) {
      const int n = n0 + k;
      const float v = (WE.e[m][n] - mean) * rs * poLg[n] + poLb[n];
      As0[m][n] = __float2bfloat16(v * sigmoidf_(v));
    }
  }
  __syncthreads();

  // ---- phase 3: po GEMM + residual
  async_copy_w(wpo, WE.w, tid);
  async_wait0();
  __syncthreads();
  v8f acc2[4];
#pragma unroll
  for (int mb = 0; mb < 4; ++mb) acc2[mb] = zero8();
  tile_gemm(As0, WE.w, lane, wv * 16, acc2);

  {
    const int n = wv * 16 + (lane & 15);
    const float pb = pob[n];
#pragma unroll
    for (int mb = 0; mb < 4; ++mb) {
      F8 u; u.v = acc2[mb];
#pragma unroll
      for (int r = 0; r < 8; ++r) {
        const int m = mb * 16 + ((lane >> 4) << 3) + r;
        if (m < nj) {
          const size_t off = (rowbase + m) * HH + n;
          const float ein = __bfloat162float(eN[off]);
          eN[off] = __float2bfloat16(ein + u.f[r] + pb);
        }
      }
    }
  }
}

// ---------------------------------------------------------------------------
// h = h + relu(LN(Uh + agg))   (one wave per node row)
// ---------------------------------------------------------------------------
__global__ __launch_bounds__(256)
void k_h_update(float* __restrict__ hbuf, const float* __restrict__ uvab,
                const float* __restrict__ agg,
                const float* __restrict__ lnHg, const float* __restrict__ lnHb)
{
  const int tid = threadIdx.x, lane = tid & 31, wv = tid >> 5;
  const int row = blockIdx.x * 8 + wv;   // < 800
  const int n0 = lane * 4;
  float v[4];
  float s = 0.f, ss = 0.f;
#pragma unroll
  for (int k = 0; k < 4; ++k) {
    v[k] = uvab[(size_t)row * HH + n0 + k] + agg[(size_t)row * HH + n0 + k];
    s += v[k]; ss += v[k] * v[k];
  }
  for (int o = 1; o < 32; o <<= 1) { s += __shfl_xor(s, o); ss += __shfl_xor(ss, o); }
  const float mean = s * (1.f / 128.f);
  const float rs = rsqrtf(ss * (1.f / 128.f) - mean * mean + EPSC);
#pragma unroll
  for (int k = 0; k < 4; ++k) {
    const int n = n0 + k;
    hbuf[(size_t)row * HH + n] += fmaxf((v[k] - mean) * rs * lnHg[n] + lnHb[n], 0.f);
  }
}

// ---------------------------------------------------------------------------
// timestep embedding MLP (B blocks of 64 threads)
// ---------------------------------------------------------------------------
__global__ void k_temb(const int* __restrict__ ts,
                       const float* __restrict__ w1, const float* __restrict__ b1,
                       const float* __restrict__ w2, const float* __restrict__ b2,
                       float* __restrict__ temb)
{
  __shared__ float t1[64];
  const int b = blockIdx.x, n = threadIdx.x;
  const float t = (float)ts[b];
  float acc = b1[n];
  for (int k = 0; k < 128; ++k) {
    const int j = k & 63;
    const float a = t * __expf(-9.2103403719761836f * (float)j * (1.f / 64.f));
    acc += ((k < 64) ? __cosf(a) : __sinf(a)) * w1[k * 64 + n];
  }
  t1[n] = fmaxf(acc, 0.f);
  __syncthreads();
  float acc2 = b2[n];
  for (int k = 0; k < 64; ++k) acc2 += t1[k] * w2[k * 64 + n];
  temb[b * 64 + n] = acc2;
}

// tlq[b,n] = tl_b[n] + sum_k relu(temb[b,k]) * tl_w[k,n]
__global__ void k_tlq(const float* __restrict__ temb, const float* __restrict__ tlw,
                      const float* __restrict__ tlb, float* __restrict__ tlq)
{
  const int t = blockIdx.x * blockDim.x + threadIdx.x;
  if (t >= BB * HH) return;
  const int b = t >> 7, n = t & 127;
  float s = tlb[n];
  for (int k = 0; k < 64; ++k) s += fmaxf(temb[b * 64 + k], 0.f) * tlw[k * 128 + n];
  tlq[b * 128 + n] = s;
}

__global__ void k_hinit(const float* __restrict__ x, const float* __restrict__ nw,
                        const float* __restrict__ nb, float* __restrict__ hbuf)
{
  const int gid = blockIdx.x * 256 + threadIdx.x;
  if (gid >= NROWS_NODE * HH) return;
  const int row = gid >> 7, n = gid & 127;
  hbuf[gid] = x[row * 2] * nw[n] + x[row * 2 + 1] * nw[128 + n] + nb[n];
}

// bf16 conversion with transpose: d[n*128+k] = bf16(s[k*128+n])
__global__ void k_cvt_t(const float* __restrict__ s, __hip_bfloat16* __restrict__ d)
{
  const int i = blockIdx.x * 256 + threadIdx.x;     // 0..16383
  const int n = i >> 7, k = i & 127;
  d[i] = __float2bfloat16(s[k * 128 + n]);
}

__global__ void k_zero(float* __restrict__ p, int n)
{
  const int i = blockIdx.x * 256 + threadIdx.x;
  if (i < n) p[i] = 0.f;
}

// GroupNorm stats over (i,j,4ch) per (b,g) via atomics
__global__ __launch_bounds__(256)
void k_gn_stats(const __hip_bfloat16* __restrict__ e, float* __restrict__ gsum,
                float* __restrict__ gsq)
{
  __shared__ float s1[256], s2[256];
  const int tid = threadIdx.x;
  const int c = blockIdx.x, g = blockIdx.y, b = blockIdx.z;
  float s = 0.f, ss = 0.f;
  for (int idx = tid; idx < 1000 * 4; idx += 256) {
    const int row = c * 1000 + (idx >> 2);
    const int ch = g * 4 + (idx & 3);
    const float v = __bfloat162float(e[((size_t)b * (VV * VV) + row) * HH + ch]);
    s += v; ss += v * v;
  }
  s1[tid] = s; s2[tid] = ss; __syncthreads();
  for (int o = 128; o > 0; o >>= 1) {
    if (tid < o) { s1[tid] += s1[tid + o]; s2[tid] += s2[tid + o]; }
    __syncthreads();
  }
  if (tid == 0) {
    unsafeAtomicAdd(&gsum[b * 32 + g], s1[0]);
    unsafeAtomicAdd(&gsq[b * 32 + g], s2[0]);
  }
}

// GroupNorm apply + relu + 1x1 conv head (one wave per edge row)
__global__ __launch_bounds__(256)
void k_head(const __hip_bfloat16* __restrict__ e, const float* __restrict__ gsum,
            const float* __restrict__ gsq, const float* __restrict__ gng,
            const float* __restrict__ gnb, const float* __restrict__ ow,
            const float* __restrict__ ob, float* __restrict__ out)
{
  const int tid = threadIdx.x, lane = tid & 31, wv = tid >> 5;
  const int rowg = blockIdx.x * 8 + wv;          // < 160000
  const int b = rowg / (VV * VV);
  const int rij = rowg - b * (VV * VV);
  const float inv = 1.f / 160000.f;
  const float mean = gsum[b * 32 + lane] * inv;            // group g == lane (4ch/group)
  const float var = gsq[b * 32 + lane] * inv - mean * mean;
  const float rs = rsqrtf(var + EPSC);
  const size_t base = (size_t)rowg * HH + lane * 4;
  float a0 = 0.f, a1 = 0.f;
#pragma unroll
  for (int k = 0; k < 4; ++k) {
    const int n = lane * 4 + k;
    const float v = __bfloat162float(e[base + k]);
    const float r = fmaxf((v - mean) * rs * gng[n] + gnb[n], 0.f);
    a0 += r * ow[n]; a1 += r * ow[128 + n];
  }
  for (int o = 1; o < 32; o <<= 1) { a0 += __shfl_xor(a0, o); a1 += __shfl_xor(a1, o); }
  if (lane == 0) {
    out[(size_t)(b * 2 + 0) * (VV * VV) + rij] = a0 + ob[0];
    out[(size_t)(b * 2 + 1) * (VV * VV) + rij] = a1 + ob[1];
  }
}

// ---------------------------------------------------------------------------
extern "C" void kernel_launch(void* const* d_in, const int* in_sizes, int n_in,
                              void* d_out, int out_size, void* d_ws, size_t ws_size,
                              hipStream_t stream)
{
  (void)in_sizes; (void)n_in; (void)out_size; (void)ws_size;
  const float* x       = (const float*)d_in[0];
  const float* gnoisy  = (const float*)d_in[1];
  const float* gcond   = (const float*)d_in[2];
  const int*   tsteps  = (const int*)  d_in[3];
  const float* node_w  = (const float*)d_in[4];
  const float* node_b  = (const float*)d_in[5];
  const float* edge_w  = (const float*)d_in[6];
  const float* edge_b  = (const float*)d_in[7];
  const float* te_w1   = (const float*)d_in[8];
  const float* te_b1   = (const float*)d_in[9];
  const float* te_w2   = (const float*)d_in[10];
  const float* te_b2   = (const float*)d_in[11];
  const float* ln_h_g  = (const float*)d_in[12];
  const float* ln_h_b  = (const float*)d_in[13];
  const float* ln_e_g  = (const float*)d_in[14];
  const float* ln_e_b  = (const float*)d_in[15];
  const float* tl_w    = (const float*)d_in[16];
  const float* tl_b    = (const float*)d_in[17];
  const float* po_ln_g = (const float*)d_in[18];
  const float* po_ln_b = (const float*)d_in[19];
  const float* po_w    = (const float*)d_in[20];
  const float* po_b    = (const float*)d_in[21];
  const float* gn_g    = (const float*)d_in[22];
  const float* gn_b    = (const float*)d_in[23];
  const float* out_w   = (const float*)d_in[24];
  const float* out_b   = (const float*)d_in[25];
  const float* U_w = (const float*)d_in[26]; const float* U_b = (const float*)d_in[27];
  const float* V_w = (const float*)d_in[28]; const float* V_b = (const float*)d_in[29];
  const float* A_w = (const float*)d_in[30]; const float* A_b = (const float*)d_in[31];
  const float* B_w = (const float*)d_in[32]; const float* B_b = (const float*)d_in[33];
  const float* C_w = (const float*)d_in[34]; const float* C_b = (const float*)d_in[35];
  const float* D_w = (const float*)d_in[36]; const float* D_b = (const float*)d_in[37];

  // -------- workspace carve-out --------
  const size_t NE = (size_t)NROWS_EDGE * HH;     // 20,480,000
  const size_t WN = (size_t)HH * HH;             // 16,384
  char* wsb = (char*)d_ws;
  size_t off = 0;
  auto take = [&](size_t bytes) -> char* {
    char* p = wsb + off; off += (bytes + 255) & ~(size_t)255; return p;
  };
  __hip_bfloat16* eN   = (__hip_bfloat16*)take(NE * 2);
  __hip_bfloat16* eC   = (__hip_bfloat16*)take(NE * 2);
  float* hbuf          = (float*)take((size_t)NROWS_NODE * HH * 4);
  float* uvab          = (float*)take((size_t)4 * NROWS_NODE * HH * 4);
  float* agg           = (float*)take((size_t)NROWS_NODE * HH * 4);
  float* temb          = (float*)take(BB * 64 * 4);
  float* tlq           = (float*)take(BB * HH * 4);
  float* gsum          = (float*)take(256 * 4);          // gsum[128] ++ gsq[128]
  float* gsq           = gsum + 128;
  __hip_bfloat16* wbf  = (__hip_bfloat16*)take((size_t)22 * WN * 2);

  // -------- weight conversion to bf16, pre-transposed [n][k] --------
  auto cvt = [&](const float* src, __hip_bfloat16* dst) {
    k_cvt_t<<<64, 256, 0, stream>>>(src, dst);
  };
  cvt(edge_w, wbf + (size_t)21 * WN);
  for (int l = 0; l < LL; ++l) {
    cvt(U_w + (size_t)l * WN, wbf + (size_t)(l * 7 + 0) * WN);
    cvt(V_w + (size_t)l * WN, wbf + (size_t)(l * 7 + 1) * WN);
    cvt(A_w + (size_t)l * WN, wbf + (size_t)(l * 7 + 2) * WN);
    cvt(B_w + (size_t)l * WN, wbf + (size_t)(l * 7 + 3) * WN);
    cvt(C_w + (size_t)l * WN, wbf + (size_t)(l * 7 + 4) * WN);
    cvt(D_w + (size_t)l * WN, wbf + (size_t)(l * 7 + 5) * WN);
    cvt(po_w + (size_t)l * WN, wbf + (size_t)(l * 7 + 6) * WN);
  }

  // -------- embeddings --------
  k_temb<<<BB, 64, 0, stream>>>(tsteps, te_w1, te_b1, te_w2, te_b2, temb);
  k_hinit<<<400, 256, 0, stream>>>(x, node_w, node_b, hbuf);
  k_edge_embed<<<NROWS_EDGE / 64, 256, 0, stream>>>(gnoisy, wbf + (size_t)21 * WN, edge_b, eN);
  k_edge_embed<<<NROWS_EDGE / 64, 256, 0, stream>>>(gcond,  wbf + (size_t)21 * WN, edge_b, eC);

  // -------- layers --------
  for (int l = 0; l < LL; ++l) {
    k_node_gemm<<<dim3(13, 4), 256, 0, stream>>>(
        hbuf, wbf + (size_t)l * 7 * WN,
        U_b + l * HH, V_b + l * HH, A_b + l * HH, B_b + l * HH, uvab);
    k_tlq<<<2, 256, 0, stream>>>(temb, tl_w + (size_t)l * 64 * HH, tl_b + l * HH, tlq);
    k_zero<<<400, 256, 0, stream>>>(agg, NROWS_NODE * HH);
    k_edge_layer<<<dim3(4, VV, BB), 256, 0, stream>>>(
        eN, eC,
        wbf + (size_t)(l * 7 + 4) * WN, wbf + (size_t)(l * 7 + 5) * WN,
        wbf + (size_t)(l * 7 + 6) * WN,
        C_b + l * HH, D_b + l * HH, po_b + l * HH,
        ln_e_g + l * HH, ln_e_b + l * HH, po_ln_g + l * HH, po_ln_b + l * HH,
        uvab, agg, tlq);
    k_h_update<<<100, 256, 0, stream>>>(hbuf, uvab, agg, ln_h_g + l * HH, ln_h_b + l * HH);
  }

  // -------- output head --------
  k_zero<<<1, 256, 0, stream>>>(gsum, 256);
  k_gn_stats<<<dim3(40, 32, BB), 256, 0, stream>>>(eN, gsum, gsq);
  k_head<<<NROWS_EDGE / 8, 256, 0, stream>>>(eN, gsum, gsq, gn_g, gn_b, out_w, out_b,
                                             (float*)d_out);
}
